// WanMMAttention_28810640621736
// MI455X (gfx1250) — matrix operand order repaired and checked by reading, not compiled, not run on hardware
//
#include <hip/hip_runtime.h>

// ---------------------------------------------------------------------------
// WanMM dual-stream attention for MI455X (gfx1250, wave32, WMMA).
// cast->bf16, WMMA GEMM projections (64x256 block tile), RMSNorm+RoPE scatter,
// flash attention with double-buffered global_load_async_to_lds_b128 K/V
// staging + WMMA QK^T / PV + online softmax, WMMA output GEMMs.
// ---------------------------------------------------------------------------

#define DIMM   2048
#define NHH    16
#define HDD    128
#define SEQ    2048
#define SCOND  512
#define STT    2560   // SCOND + SEQ

#define KROW   136    // padded LDS row stride (elements) for K tile (32 x 128)
#define VROW   40     // padded LDS row stride (elements) for V tile (128 x 32)

typedef __attribute__((ext_vector_type(16))) __bf16 v16bf;
typedef __attribute__((ext_vector_type(8)))  float  v8f;

union FragAB { v16bf v; uint4 q[2]; };

__device__ __forceinline__ unsigned short f32_to_bf16(float f) {
  unsigned int u = __float_as_uint(f);
  u += 0x7FFFu + ((u >> 16) & 1u);          // round to nearest even
  return (unsigned short)(u >> 16);
}

// CDNA5 async copy: one instruction moves 16B per active lane into LDS.
__device__ __forceinline__ void async_load_b128(unsigned int lds_byte_addr,
                                                const void* gptr) {
  asm volatile("global_load_async_to_lds_b128 %0, %1, off"
               :: "v"(lds_byte_addr), "v"((unsigned long long)(uintptr_t)gptr)
               : "memory");
}
__device__ __forceinline__ void wait_async0() {
  asm volatile("s_wait_asynccnt 0x0" ::: "memory");
}

// --------------------------- cast fp32 -> bf16 -----------------------------
__global__ __launch_bounds__(256) void cast_f32_bf16(const float* __restrict__ in,
                                                     unsigned short* __restrict__ out,
                                                     int n) {
  int i = blockIdx.x * 256 + threadIdx.x;
  int stride = gridDim.x * 256;
  for (; i < n; i += stride) out[i] = f32_to_bf16(in[i]);
}

// ------------------------------- WMMA GEMM ---------------------------------
// C[m,n] = sum_k A[m,k] * W[n,k] + bias[n]   (A,W bf16 row-major; C fp32)
// Block tile 64(M) x 256(N); 8 waves arranged 2x4; wave tile 32x64 = 2x4 WMMA.
__global__ __launch_bounds__(256) void gemm_bf16_f32(const unsigned short* __restrict__ A,
                                                     const unsigned short* __restrict__ W,
                                                     const float* __restrict__ bias,
                                                     float* __restrict__ C,
                                                     int M, int N, int K) {
  const int wave = threadIdx.x >> 5;
  const int lane = threadIdx.x & 31;
  const int wm = wave >> 2, wn = wave & 3;
  const int m0 = blockIdx.y * 64 + wm * 32;
  const int n0 = blockIdx.x * 256 + wn * 64;
  const int hi = lane >> 4, ln = lane & 15;

  v8f acc[2][4] = {{{}, {}, {}, {}}, {{}, {}, {}, {}}};

  // A fragment: lane row = m; elems 0..7 = K[hi*8 .. +7], 8..15 = K[hi*8+16 .. +23]
  const unsigned short* Ap[2];
  Ap[0] = A + (size_t)(m0 + ln) * K + hi * 8;
  Ap[1] = Ap[0] + (size_t)16 * K;
  // B fragment: lane col = n; elems 0..7 = K[hi*8 .. +7], 8..15 = K[hi*8+16 .. +23]
  const unsigned short* Wp[4];
  #pragma unroll
  for (int j = 0; j < 4; ++j) Wp[j] = W + (size_t)(n0 + j * 16 + ln) * K + hi * 8;

  for (int k0 = 0; k0 < K; k0 += 32) {
    if (k0 + 32 < K) {
      __builtin_prefetch(Ap[0] + k0 + 32, 0, 3);
      __builtin_prefetch(Wp[0] + k0 + 32, 0, 3);
      __builtin_prefetch(Wp[2] + k0 + 32, 0, 3);
    }
    FragAB a[2], b[4];
    #pragma unroll
    for (int i = 0; i < 2; ++i) {
      a[i].q[0] = *(const uint4*)(Ap[i] + k0);
      a[i].q[1] = *(const uint4*)(Ap[i] + k0 + 16);
    }
    #pragma unroll
    for (int j = 0; j < 4; ++j) {
      b[j].q[0] = *(const uint4*)(Wp[j] + k0);
      b[j].q[1] = *(const uint4*)(Wp[j] + k0 + 16);
    }
    #pragma unroll
    for (int i = 0; i < 2; ++i)
      #pragma unroll
      for (int j = 0; j < 4; ++j)
        acc[i][j] = __builtin_amdgcn_wmma_f32_16x16x32_bf16(false, a[i].v, false, b[j].v,
                                                            (short)0, acc[i][j], false, false);
  }

  // C layout: lane col = ln; VGPR r -> row r + 8*hi
  #pragma unroll
  for (int j = 0; j < 4; ++j) {
    const int n = n0 + j * 16 + ln;
    const float bv = bias[n];
    #pragma unroll
    for (int i = 0; i < 2; ++i)
      #pragma unroll
      for (int r = 0; r < 8; ++r)
        C[(size_t)(m0 + i * 16 + r + 8 * hi) * N + n] = acc[i][j][r] + bv;
  }
}

// ------------------- RMSNorm (over DIM) + RoPE + scatter -------------------
// dst layout: (NH, ST, HD) bf16; one block per sequence row.
__global__ __launch_bounds__(256) void rmsnorm_rope_scatter(const float* __restrict__ X,
                                                            const float* __restrict__ nw,
                                                            const float* __restrict__ cosv,
                                                            const float* __restrict__ sinv,
                                                            unsigned short* __restrict__ dst,
                                                            int seqOff) {
  const int s = blockIdx.x;
  const float* x = X + (size_t)s * DIMM;
  __shared__ float red[256];
  float ss = 0.f;
  for (int i = threadIdx.x; i < DIMM; i += 256) { float v = x[i]; ss += v * v; }
  red[threadIdx.x] = ss;
  __syncthreads();
  for (int off = 128; off > 0; off >>= 1) {
    if (threadIdx.x < off) red[threadIdx.x] += red[threadIdx.x + off];
    __syncthreads();
  }
  const float scale = rsqrtf(red[0] * (1.0f / DIMM) + 1e-6f);
  const float* crow = cosv + (size_t)s * (HDD / 2);
  const float* srow = sinv + (size_t)s * (HDD / 2);
  for (int p = threadIdx.x; p < DIMM / 2; p += 256) {
    const int head = p >> 6;          // p / (HD/2)
    const int j    = p & 63;
    const float c = crow[j], sn = srow[j];
    const float x0 = x[2 * p]     * scale * nw[2 * p];
    const float x1 = x[2 * p + 1] * scale * nw[2 * p + 1];
    const size_t base = ((size_t)head * STT + (seqOff + s)) * HDD + 2 * j;
    dst[base]     = f32_to_bf16(x0 * c  - x1 * sn);
    dst[base + 1] = f32_to_bf16(x0 * sn + x1 * c);
  }
}

// ------------------ V scatter into transposed (NH, HD, ST) -----------------
__global__ __launch_bounds__(256) void v_scatter(const float* __restrict__ X,
                                                 unsigned short* __restrict__ Vt,
                                                 int seqOff) {
  const int s = blockIdx.x;
  const float* x = X + (size_t)s * DIMM;
  for (int e = threadIdx.x; e < DIMM; e += 256) {
    const int head = e >> 7, d = e & 127;
    Vt[((size_t)head * HDD + d) * STT + seqOff + s] = f32_to_bf16(x[e]);
  }
}

// ------------------------------ Flash attention ----------------------------
// Grid: (NH, ST/64). Block: 128 threads = 4 waves; each wave owns 16 q rows.
// K/V tiles are shared by all 4 waves -> double-buffered async-to-LDS staging.
// Qj,Kj: (NH, ST, HD) bf16.  Vt: (NH, HD, ST) bf16.  O: (ST, DIM) bf16.
__global__ __launch_bounds__(128) void flash_attn(const unsigned short* __restrict__ Qj,
                                                  const unsigned short* __restrict__ Kj,
                                                  const unsigned short* __restrict__ Vt,
                                                  unsigned short* __restrict__ O,
                                                  const int* __restrict__ seq_lens) {
  const int head = blockIdx.x;
  const int tid  = threadIdx.x;
  const int wave = tid >> 5;
  const int lane = tid & 31;
  const int hi = lane >> 4, ln = lane & 15;
  const int qb = blockIdx.y * 64 + wave * 16;
  const int seq_len = seq_lens[0];
  const float scale = 0.08838834764831845f;   // 1/sqrt(128)

  __shared__ alignas(16) unsigned short ktile[2][32 * KROW];   // 32 kv rows x HD
  __shared__ alignas(16) unsigned short vtile[2][HDD * VROW];  // HD rows x 32 kv
  __shared__ alignas(16) unsigned short plds[4][16][40];       // per-wave P tile

  // Q A-fragments for 4 chunks of the HD=128 contraction (loaded once)
  FragAB qf[4];
  const unsigned short* qrow = Qj + ((size_t)head * STT + qb + ln) * HDD + hi * 8;
  #pragma unroll
  for (int dc = 0; dc < 4; ++dc) {
    qf[dc].q[0] = *(const uint4*)(qrow + dc * 32);
    qf[dc].q[1] = *(const uint4*)(qrow + dc * 32 + 16);
  }

  // Async stage of one 32-wide kv tile into LDS buffer `buf`.
  const int kr = tid >> 2;         // K tile row handled by this thread (0..31)
  const int kq = (tid & 3) * 32;   // element offset within the K row
  auto issue_tile = [&](int buf, int kv) {
    const unsigned short* kg = Kj + ((size_t)head * STT + kv + kr) * HDD + kq;
    unsigned int klds = (unsigned int)(uintptr_t)&ktile[buf][kr * KROW + kq];
    #pragma unroll
    for (int j = 0; j < 4; ++j)
      async_load_b128(klds + j * 16, kg + j * 8);
    const unsigned short* vg = Vt + ((size_t)head * HDD + tid) * STT + kv;
    unsigned int vlds = (unsigned int)(uintptr_t)&vtile[buf][tid * VROW];
    #pragma unroll
    for (int j = 0; j < 4; ++j)
      async_load_b128(vlds + j * 16, vg + j * 8);
  };

  v8f acc[8] = {{}, {}, {}, {}, {}, {}, {}, {}};
  float mrow[8], lrow[8];
  #pragma unroll
  for (int r = 0; r < 8; ++r) { mrow[r] = -1e30f; lrow[r] = 0.f; }

  issue_tile(0, 0);
  int cur = 0;

  for (int kv0 = 0; kv0 < STT; kv0 += 32) {
    wait_async0();        // my loads for buf[cur] have landed in LDS
    __syncthreads();      // everyone's loads landed; prev reads of buf[cur^1] done
    if (kv0 + 32 < STT) issue_tile(cur ^ 1, kv0 + 32);

    // ---- S = Q K^T for 16 x 32 tile (two 16x16 C tiles), K from LDS ----
    v8f s01[2] = {{}, {}};
    #pragma unroll
    for (int t = 0; t < 2; ++t) {
      const unsigned short* kp = &ktile[cur][(t * 16 + ln) * KROW + hi * 8];
      #pragma unroll
      for (int dc = 0; dc < 4; ++dc) {
        FragAB kf;
        kf.q[0] = *(const uint4*)(kp + dc * 32);
        kf.q[1] = *(const uint4*)(kp + dc * 32 + 16);
        s01[t] = __builtin_amdgcn_wmma_f32_16x16x32_bf16(false, qf[dc].v, false, kf.v,
                                                         (short)0, s01[t], false, false);
      }
    }
    // ---- scale + key mask (per-lane column) ----
    const float mk0 = (kv0 + ln      < seq_len) ? 0.f : -1e30f;
    const float mk1 = (kv0 + 16 + ln < seq_len) ? 0.f : -1e30f;
    float rmax[8];
    #pragma unroll
    for (int r = 0; r < 8; ++r) {
      s01[0][r] = s01[0][r] * scale + mk0;
      s01[1][r] = s01[1][r] * scale + mk1;
      rmax[r] = fmaxf(s01[0][r], s01[1][r]);
    }
    #pragma unroll
    for (int off = 8; off > 0; off >>= 1)
      #pragma unroll
      for (int r = 0; r < 8; ++r) rmax[r] = fmaxf(rmax[r], __shfl_xor(rmax[r], off, 32));
    // ---- online softmax update ----
    float alpha[8], rs[8];
    #pragma unroll
    for (int r = 0; r < 8; ++r) {
      const float mn = fmaxf(mrow[r], rmax[r]);
      alpha[r] = __expf(mrow[r] - mn);
      mrow[r] = mn;
      s01[0][r] = __expf(s01[0][r] - mn);
      s01[1][r] = __expf(s01[1][r] - mn);
      rs[r] = s01[0][r] + s01[1][r];
    }
    #pragma unroll
    for (int off = 8; off > 0; off >>= 1)
      #pragma unroll
      for (int r = 0; r < 8; ++r) rs[r] += __shfl_xor(rs[r], off, 32);
    #pragma unroll
    for (int r = 0; r < 8; ++r) lrow[r] = lrow[r] * alpha[r] + rs[r];
    #pragma unroll
    for (int dt = 0; dt < 8; ++dt)
      #pragma unroll
      for (int r = 0; r < 8; ++r) acc[dt][r] *= alpha[r];
    // ---- redistribute P: C layout -> A layout through per-wave LDS ----
    #pragma unroll
    for (int r = 0; r < 8; ++r) {
      const int row = r + 8 * hi;
      plds[wave][row][ln]      = f32_to_bf16(s01[0][r]);
      plds[wave][row][16 + ln] = f32_to_bf16(s01[1][r]);
    }
    FragAB pf;
    const unsigned short* pb = &plds[wave][ln][hi * 8];
    pf.q[0] = *(const uint4*)pb;
    pf.q[1] = *(const uint4*)(pb + 16);
    // ---- O += P V (contraction over 32 kv rows; V columns from LDS) ----
    #pragma unroll
    for (int dt = 0; dt < 8; ++dt) {
      const unsigned short* vp = &vtile[cur][(dt * 16 + ln) * VROW + hi * 8];
      FragAB vf;
      vf.q[0] = *(const uint4*)vp;
      vf.q[1] = *(const uint4*)(vp + 16);
      acc[dt] = __builtin_amdgcn_wmma_f32_16x16x32_bf16(false, pf.v, false, vf.v,
                                                        (short)0, acc[dt], false, false);
    }
    cur ^= 1;
  }

  // ---- epilogue: normalize and store bf16 into (ST, DIM) ----
  float invl[8];
  #pragma unroll
  for (int r = 0; r < 8; ++r) invl[r] = 1.0f / lrow[r];
  #pragma unroll
  for (int dt = 0; dt < 8; ++dt)
    #pragma unroll
    for (int r = 0; r < 8; ++r)
      O[(size_t)(qb + r + 8 * hi) * DIMM + head * HDD + dt * 16 + ln] =
          f32_to_bf16(acc[dt][r] * invl[r]);
}

// ---------------------------------------------------------------------------
extern "C" void kernel_launch(void* const* d_in, const int* in_sizes, int n_in,
                              void* d_out, int out_size, void* d_ws, size_t ws_size,
                              hipStream_t stream) {
  (void)in_sizes; (void)n_in; (void)out_size; (void)ws_size;
  char* ws = (char*)d_ws;
  size_t off = 0;
  auto alloc = [&](size_t bytes) {
    char* p = ws + off;
    off += (bytes + 255) & ~(size_t)255;
    return p;
  };
  // workspace (~131 MB total)
  unsigned short* wbf[8];
  for (int i = 0; i < 8; ++i) wbf[i] = (unsigned short*)alloc((size_t)DIMM * DIMM * 2);
  unsigned short* xbf  = (unsigned short*)alloc((size_t)SEQ   * DIMM * 2);
  unsigned short* xcbf = (unsigned short*)alloc((size_t)SCOND * DIMM * 2);
  float*          tmp  = (float*)         alloc((size_t)SEQ   * DIMM * 4);
  unsigned short* Qj   = (unsigned short*)alloc((size_t)NHH * STT * HDD * 2);
  unsigned short* Kj   = (unsigned short*)alloc((size_t)NHH * STT * HDD * 2);
  unsigned short* Vt   = (unsigned short*)alloc((size_t)NHH * HDD * STT * 2);
  unsigned short* AO   = (unsigned short*)alloc((size_t)STT * DIMM * 2);

  const float* x        = (const float*)d_in[0];
  const float* xc       = (const float*)d_in[1];
  const int*   seq_lens = (const int*)  d_in[2];
  const float* fcos     = (const float*)d_in[3];
  const float* fsin     = (const float*)d_in[4];
  const float* fccos    = (const float*)d_in[5];
  const float* fcsin    = (const float*)d_in[6];
  const float* Wm[8]; for (int i = 0; i < 8; ++i) Wm[i] = (const float*)d_in[7 + i];   // qw kw vw ow q2w k2w v2w o2w
  const float* Bb[8]; for (int i = 0; i < 8; ++i) Bb[i] = (const float*)d_in[15 + i];  // qb kb vb ob q2b k2b v2b o2b
  const float* nqw  = (const float*)d_in[23];
  const float* nkw  = (const float*)d_in[24];
  const float* nq2w = (const float*)d_in[25];
  const float* nk2w = (const float*)d_in[26];

  // 1) casts to bf16
  for (int i = 0; i < 8; ++i)
    cast_f32_bf16<<<1024, 256, 0, stream>>>(Wm[i], wbf[i], DIMM * DIMM);
  cast_f32_bf16<<<1024, 256, 0, stream>>>(x,  xbf,  SEQ * DIMM);
  cast_f32_bf16<<<512,  256, 0, stream>>>(xc, xcbf, SCOND * DIMM);

  const dim3 gX(DIMM / 256, SEQ / 64);    // (8,32) for M=2048
  const dim3 gC(DIMM / 256, SCOND / 64);  // (8,8)  for M=512

  // 2) x-stream projections + norm/rope/scatter (joint seq offset = SCOND)
  gemm_bf16_f32<<<gX, 256, 0, stream>>>(xbf, wbf[0], Bb[0], tmp, SEQ, DIMM, DIMM);
  rmsnorm_rope_scatter<<<SEQ, 256, 0, stream>>>(tmp, nqw, fcos, fsin, Qj, SCOND);
  gemm_bf16_f32<<<gX, 256, 0, stream>>>(xbf, wbf[1], Bb[1], tmp, SEQ, DIMM, DIMM);
  rmsnorm_rope_scatter<<<SEQ, 256, 0, stream>>>(tmp, nkw, fcos, fsin, Kj, SCOND);
  gemm_bf16_f32<<<gX, 256, 0, stream>>>(xbf, wbf[2], Bb[2], tmp, SEQ, DIMM, DIMM);
  v_scatter<<<SEQ, 256, 0, stream>>>(tmp, Vt, SCOND);

  // 3) cond-stream projections (joint seq offset = 0)
  gemm_bf16_f32<<<gC, 256, 0, stream>>>(xcbf, wbf[4], Bb[4], tmp, SCOND, DIMM, DIMM);
  rmsnorm_rope_scatter<<<SCOND, 256, 0, stream>>>(tmp, nq2w, fccos, fcsin, Qj, 0);
  gemm_bf16_f32<<<gC, 256, 0, stream>>>(xcbf, wbf[5], Bb[5], tmp, SCOND, DIMM, DIMM);
  rmsnorm_rope_scatter<<<SCOND, 256, 0, stream>>>(tmp, nk2w, fccos, fcsin, Kj, 0);
  gemm_bf16_f32<<<gC, 256, 0, stream>>>(xcbf, wbf[6], Bb[6], tmp, SCOND, DIMM, DIMM);
  v_scatter<<<SCOND, 256, 0, stream>>>(tmp, Vt, 0);

  // 4) joint flash attention
  flash_attn<<<dim3(NHH, STT / 64), 128, 0, stream>>>(Qj, Kj, Vt, AO, seq_lens);

  // 5) output projections straight into d_out (x part first, then cond part)
  float* out = (float*)d_out;
  gemm_bf16_f32<<<gX, 256, 0, stream>>>(AO + (size_t)SCOND * DIMM, wbf[3], Bb[3],
                                        out, SEQ, DIMM, DIMM);
  gemm_bf16_f32<<<gC, 256, 0, stream>>>(AO, wbf[7], Bb[7],
                                        out + (size_t)SEQ * DIMM, SCOND, DIMM, DIMM);
}
